// CausalSelfAttention_7516192768804
// MI455X (gfx1250) — compile-verified, matrix operand order checked by reading, not run on hardware
//
#include <hip/hip_runtime.h>
#include <hip/hip_bf16.h>

#define D_MODEL 1024
#define NHEADS  16
#define DKDIM   64
#define BATCH   2
#define SEQ     2048
#define NQKV    3072
#define ROWS    (BATCH * SEQ)   // 4096

typedef _Float16 half16 __attribute__((ext_vector_type(16)));
typedef float    float8 __attribute__((ext_vector_type(8)));

// K-offset of the f16 pair held in fragment register v for lane-half h
// (CDNA5 ISA 7.12.2, 16-bit A-matrix 16x32 layout)
__device__ __forceinline__ int kb_off(int v, int h) {
    return ((v >= 4) ? 16 : 0) + h * 8 + (v & 3) * 2;
}

// A fragment: 16 rows x 32 K, element (m,k) = src[(row0+m)*ld + k0+k]
__device__ __forceinline__ half16 ld_frag_rm(const _Float16* src, int ld, int row0, int k0) {
    int lane = threadIdx.x & 31;
    int r = row0 + (lane & 15);
    int h = lane >> 4;
    const _Float16* p = src + (size_t)r * ld + k0;
    half16 f;
#pragma unroll
    for (int v = 0; v < 8; ++v) {
        int kb = kb_off(v, h);
        f[2 * v]     = p[kb];
        f[2 * v + 1] = p[kb + 1];
    }
    return f;
}

// B fragment: 32 K x 16 N, element (k,n) = src[(n0+n)*ld + k0+k]  (B stored N-major)
__device__ __forceinline__ half16 ld_fragB_t(const _Float16* src, int ld, int k0, int n0) {
    int lane = threadIdx.x & 31;
    int n = n0 + (lane & 15);
    int h = lane >> 4;
    const _Float16* p = src + (size_t)n * ld + k0;
    half16 f;
#pragma unroll
    for (int v = 0; v < 8; ++v) {
        int kb = kb_off(v, h);
        f[2 * v]     = p[kb];
        f[2 * v + 1] = p[kb + 1];
    }
    return f;
}

// B fragment: 32 K x 16 N, element (k,n) = src[(k0+k)*ld + n0+n]  (B stored K-major)
__device__ __forceinline__ half16 ld_fragB_rm(const _Float16* src, int ld, int k0, int n0) {
    int lane = threadIdx.x & 31;
    int n = n0 + (lane & 15);
    int h = lane >> 4;
    half16 f;
#pragma unroll
    for (int v = 0; v < 8; ++v) {
        int kb = k0 + kb_off(v, h);
        f[2 * v]     = src[(size_t)kb * ld + n];
        f[2 * v + 1] = src[(size_t)(kb + 1) * ld + n];
    }
    return f;
}

__device__ __forceinline__ float8 wmma_f16(half16 a, half16 b, float8 c) {
    return __builtin_amdgcn_wmma_f32_16x16x32_f16(false, a, false, b, (short)0, c, false, false);
}

// ---------------- conversion kernels ----------------
__global__ void k_cvt(const float* __restrict__ in, _Float16* __restrict__ out, int n) {
    int i = blockIdx.x * blockDim.x + threadIdx.x;
    if (i < n) out[i] = (_Float16)in[i];
}

// in: [K x N] f32 row-major -> out: [N x K] f16 (transposed)
__global__ void k_cvt_t(const float* __restrict__ in, _Float16* __restrict__ out, int K, int N) {
    int i = blockIdx.x * blockDim.x + threadIdx.x;
    if (i < K * N) {
        int n = i / K;
        int k = i - n * K;
        out[i] = (_Float16)in[(size_t)k * N + n];
    }
}

// ---------------- QKV projection GEMM ----------------
// 32x64 register block per wave: 2 A-frags x 4 B-frags -> 8 WMMAs per k-step.
// Output scattered into Q/K/V head-split f16 tensors [B,H,S,64].
__global__ void __launch_bounds__(128)
k_gemm_qkv(const _Float16* __restrict__ Xh, const _Float16* __restrict__ Wt,
           const float* __restrict__ bias,
           _Float16* __restrict__ Qh, _Float16* __restrict__ Kh, _Float16* __restrict__ Vh) {
    int wave = threadIdx.x >> 5;
    int i0 = blockIdx.x * 32;
    int j0 = (blockIdx.y * 4 + wave) * 64;
    float8 c[2][4] = {};
    for (int kk = 0; kk < D_MODEL; kk += 32) {
        half16 a0 = ld_frag_rm(Xh, D_MODEL, i0,      kk);
        half16 a1 = ld_frag_rm(Xh, D_MODEL, i0 + 16, kk);
#pragma unroll
        for (int t = 0; t < 4; ++t) {
            half16 b = ld_fragB_t(Wt, D_MODEL, kk, j0 + 16 * t);
            c[0][t] = wmma_f16(a0, b, c[0][t]);
            c[1][t] = wmma_f16(a1, b, c[1][t]);
        }
    }
    int lane = threadIdx.x & 31;
    int n = lane & 15, h = lane >> 4;
    // j0 is a multiple of 64 -> whole 64-col block lies in one head of one of Q/K/V
    int which = j0 >> 10;                 // 0=Q 1=K 2=V
    int head  = (j0 & (D_MODEL - 1)) >> 6;
    _Float16* dst = (which == 0) ? Qh : (which == 1) ? Kh : Vh;
#pragma unroll
    for (int t = 0; t < 4; ++t) {
        int j  = j0 + 16 * t + n;
        int dk = 16 * t + n;
        float bj = bias[j];
#pragma unroll
        for (int rt = 0; rt < 2; ++rt) {
#pragma unroll
            for (int v = 0; v < 8; ++v) {
                int i  = i0 + 16 * rt + v + 8 * h;
                int bb = i >> 11;           // / SEQ
                int s  = i & (SEQ - 1);
                dst[(((size_t)bb * NHEADS + head) * SEQ + s) * DKDIM + dk] =
                    (_Float16)(c[rt][t][v] + bj);
            }
        }
    }
}

// ---------------- flash attention (one wave per 32-query tile) ----------------
__global__ void __launch_bounds__(32)
k_attn(const _Float16* __restrict__ Qh, const _Float16* __restrict__ Kh,
       const _Float16* __restrict__ Vh, _Float16* __restrict__ Attnh) {
    __shared__ _Float16 P[32][34];

    int bid = blockIdx.x;
    int qt  = bid & (SEQ / 32 - 1);
    int bh  = bid >> 6;                 // SEQ/32 == 64
    int q0  = qt * 32;
    const _Float16* Q  = Qh + (size_t)bh * SEQ * DKDIM;
    const _Float16* Ks = Kh + (size_t)bh * SEQ * DKDIM;
    const _Float16* Vs = Vh + (size_t)bh * SEQ * DKDIM;

    int lane = threadIdx.x & 31;
    int n = lane & 15, h = lane >> 4;

    half16 aq[2][2];
#pragma unroll
    for (int rt = 0; rt < 2; ++rt)
#pragma unroll
        for (int kc = 0; kc < 2; ++kc)
            aq[rt][kc] = ld_frag_rm(Q, DKDIM, q0 + 16 * rt, 32 * kc);

    float8 o[2][4] = {};
    float mrow[2][8], lrow[2][8];
#pragma unroll
    for (int rt = 0; rt < 2; ++rt)
#pragma unroll
        for (int v = 0; v < 8; ++v) { mrow[rt][v] = -__builtin_inff(); lrow[rt][v] = 0.f; }

    const float scale = 0.125f;  // 1/sqrt(64)

    for (int kv0 = 0; kv0 < q0 + 32; kv0 += 32) {
        // prefetch next kv tile's K and V rows (one cacheline per lane)
        if (kv0 + 32 < q0 + 32) {
            __builtin_prefetch(Ks + (size_t)(kv0 + 32 + lane) * DKDIM, 0, 0);
            __builtin_prefetch(Vs + (size_t)(kv0 + 32 + lane) * DKDIM, 0, 0);
        }
        // scores: both query row-tiles share each K fragment
        float8 s[2][2] = {};
#pragma unroll
        for (int sub = 0; sub < 2; ++sub) {
#pragma unroll
            for (int kc = 0; kc < 2; ++kc) {
                half16 bk = ld_fragB_t(Ks, DKDIM, 32 * kc, kv0 + 16 * sub);
                s[0][sub] = wmma_f16(aq[0][kc], bk, s[0][sub]);
                s[1][sub] = wmma_f16(aq[1][kc], bk, s[1][sub]);
            }
        }
#pragma unroll
        for (int rt = 0; rt < 2; ++rt) {
#pragma unroll
            for (int v = 0; v < 8; ++v) {
                int q = q0 + 16 * rt + v + 8 * h;
                float x0 = s[rt][0][v] * scale;
                float x1 = s[rt][1][v] * scale;
                if (kv0 + n > q)      x0 = -__builtin_inff();
                if (kv0 + 16 + n > q) x1 = -__builtin_inff();
                s[rt][0][v] = x0; s[rt][1][v] = x1;
            }
#pragma unroll
            for (int v = 0; v < 8; ++v) {
                // row max across the 16 lanes of this half (stays within half)
                float r = fmaxf(s[rt][0][v], s[rt][1][v]);
                r = fmaxf(r, __shfl_xor(r, 1, 32));
                r = fmaxf(r, __shfl_xor(r, 2, 32));
                r = fmaxf(r, __shfl_xor(r, 4, 32));
                r = fmaxf(r, __shfl_xor(r, 8, 32));
                float mo = mrow[rt][v];
                float mn = fmaxf(mo, r);
                float alpha = __expf(mo - mn);     // exp(-inf)=0 on first tile
                float p0 = __expf(s[rt][0][v] - mn);
                float p1 = __expf(s[rt][1][v] - mn);
                float rs = p0 + p1;
                rs += __shfl_xor(rs, 1, 32);
                rs += __shfl_xor(rs, 2, 32);
                rs += __shfl_xor(rs, 4, 32);
                rs += __shfl_xor(rs, 8, 32);
                lrow[rt][v] = lrow[rt][v] * alpha + rs;
                mrow[rt][v] = mn;
#pragma unroll
                for (int t = 0; t < 4; ++t) o[rt][t][v] *= alpha;
                P[16 * rt + v + 8 * h][n]      = (_Float16)p0;
                P[16 * rt + v + 8 * h][16 + n] = (_Float16)p1;
            }
        }
        __syncthreads();
        half16 ap0 = ld_frag_rm(&P[0][0], 34, 0,  0);  // C-layout -> A-layout via LDS
        half16 ap1 = ld_frag_rm(&P[0][0], 34, 16, 0);
        __syncthreads();
#pragma unroll
        for (int t = 0; t < 4; ++t) {
            half16 bv = ld_fragB_rm(Vs, DKDIM, kv0, 16 * t);
            o[0][t] = wmma_f16(ap0, bv, o[0][t]);
            o[1][t] = wmma_f16(ap1, bv, o[1][t]);
        }
    }

    int b = bh >> 4, hd = bh & 15;
#pragma unroll
    for (int rt = 0; rt < 2; ++rt) {
#pragma unroll
        for (int v = 0; v < 8; ++v) {
            float inv = 1.0f / lrow[rt][v];   // causal: every row has >=1 valid key
            int s = q0 + 16 * rt + v + 8 * h;
            size_t base = ((size_t)b * SEQ + s) * D_MODEL + hd * DKDIM + n;
            Attnh[base + 0]  = (_Float16)(o[rt][0][v] * inv);
            Attnh[base + 16] = (_Float16)(o[rt][1][v] * inv);
            Attnh[base + 32] = (_Float16)(o[rt][2][v] * inv);
            Attnh[base + 48] = (_Float16)(o[rt][3][v] * inv);
        }
    }
}

// ---------------- output projection GEMM (f32 out), 32x64 block per wave ----------------
__global__ void __launch_bounds__(128)
k_gemm_out(const _Float16* __restrict__ Ah, const _Float16* __restrict__ Wt,
           const float* __restrict__ bias, float* __restrict__ out) {
    int wave = threadIdx.x >> 5;
    int i0 = blockIdx.x * 32;
    int j0 = (blockIdx.y * 4 + wave) * 64;
    float8 c[2][4] = {};
    for (int kk = 0; kk < D_MODEL; kk += 32) {
        half16 a0 = ld_frag_rm(Ah, D_MODEL, i0,      kk);
        half16 a1 = ld_frag_rm(Ah, D_MODEL, i0 + 16, kk);
#pragma unroll
        for (int t = 0; t < 4; ++t) {
            half16 b = ld_fragB_t(Wt, D_MODEL, kk, j0 + 16 * t);
            c[0][t] = wmma_f16(a0, b, c[0][t]);
            c[1][t] = wmma_f16(a1, b, c[1][t]);
        }
    }
    int lane = threadIdx.x & 31;
    int n = lane & 15, h = lane >> 4;
#pragma unroll
    for (int t = 0; t < 4; ++t) {
        int j = j0 + 16 * t + n;
        float bj = bias[j];
#pragma unroll
        for (int rt = 0; rt < 2; ++rt) {
#pragma unroll
            for (int v = 0; v < 8; ++v) {
                int i = i0 + 16 * rt + v + 8 * h;
                out[(size_t)i * D_MODEL + j] = c[rt][t][v] + bj;
            }
        }
    }
}

extern "C" void kernel_launch(void* const* d_in, const int* in_sizes, int n_in,
                              void* d_out, int out_size, void* d_ws, size_t ws_size,
                              hipStream_t stream) {
    (void)in_sizes; (void)n_in; (void)out_size; (void)ws_size;
    const float* x     = (const float*)d_in[0];
    /* d_in[1] = causal mask (int) — handled analytically */
    const float* w_qkv = (const float*)d_in[2];
    const float* b_qkv = (const float*)d_in[3];
    const float* w_out = (const float*)d_in[4];
    const float* b_out = (const float*)d_in[5];

    // workspace carve-out (f16 element offsets), total 48 MB
    _Float16* base  = (_Float16*)d_ws;
    _Float16* Xh    = base;                    // 4096*1024
    _Float16* Wqkvt = base + 4194304ull;       // 3072*1024 (transposed)
    _Float16* Woutt = base + 7340032ull;       // 1024*1024 (transposed)
    _Float16* Qh    = base + 8388608ull;       // [B,H,S,64]
    _Float16* Kh    = base + 12582912ull;
    _Float16* Vh    = base + 16777216ull;
    _Float16* Ah    = base + 20971520ull;      // [B,S,1024] attn (heads concat)

    k_cvt<<<(ROWS * D_MODEL) / 256, 256, 0, stream>>>(x, Xh, ROWS * D_MODEL);
    k_cvt_t<<<(D_MODEL * NQKV) / 256, 256, 0, stream>>>(w_qkv, Wqkvt, D_MODEL, NQKV);
    k_cvt_t<<<(D_MODEL * D_MODEL) / 256, 256, 0, stream>>>(w_out, Woutt, D_MODEL, D_MODEL);

    dim3 g1(ROWS / 32, NQKV / 256);
    k_gemm_qkv<<<g1, 128, 0, stream>>>(Xh, Wqkvt, b_qkv, Qh, Kh, Vh);

    k_attn<<<BATCH * NHEADS * (SEQ / 32), 32, 0, stream>>>(Qh, Kh, Vh, Ah);

    dim3 g2(ROWS / 32, D_MODEL / 256);
    k_gemm_out<<<g2, 128, 0, stream>>>(Ah, Woutt, b_out, (float*)d_out);
}
